// AttentionJKNETGAT_36352603193550
// MI455X (gfx1250) — matrix operand order repaired and checked by reading, not compiled, not run on hardware
//
#include <hip/hip_runtime.h>
#include <hip/hip_bf16.h>
#include <math.h>
#include <stdint.h>

// ---------------- problem constants (from reference) ----------------
#define N_NODES 50000
#define E_EDGES 320000
#define DIN     128
#define DMODEL  256
#define HEADS   4
#define NLAYERS 2

// ---------------- vector types ----------------
typedef __attribute__((ext_vector_type(16))) _Float16 v16h;
typedef __attribute__((ext_vector_type(8)))  _Float16 v8h;
typedef __attribute__((ext_vector_type(4)))  _Float16 v4h;
typedef __attribute__((ext_vector_type(2)))  _Float16 v2h;
typedef __attribute__((ext_vector_type(8)))  float    v8f;

static __device__ __forceinline__ v16h cat8(v8h lo, v8h hi) {
  return __builtin_shufflevector(lo, hi, 0,1,2,3,4,5,6,7,8,9,10,11,12,13,14,15);
}

static __device__ __forceinline__ float wave_sum(float v) {
#pragma unroll
  for (int off = 16; off > 0; off >>= 1) v += __shfl_xor(v, off, 32);
  return v;
}

static __device__ __forceinline__ void atomicMaxF(float* addr, float val) {
  if (val >= 0.0f) atomicMax((int*)addr, __float_as_int(val));
  else             atomicMin((unsigned int*)addr, (unsigned int)__float_as_int(val));
}

// ---------------- async global->LDS copy (ASYNCcnt-tracked) ----------------
// generic pointers to __shared__ carry the LDS byte offset in their low 32 bits
#define USE_ASYNC_COPY 1
#if USE_ASYNC_COPY
static __device__ __forceinline__ void async_copy16(void* lds_dst, const void* gsrc) {
  unsigned ldslo = (unsigned)(uintptr_t)lds_dst;
  asm volatile("global_load_async_to_lds_b128 %0, %1, off"
               :: "v"(ldslo), "v"(gsrc) : "memory");
}
// wait until at most `n` async ops remain (async loads complete in order)
#define ASYNC_WAIT(n) asm volatile("s_wait_asynccnt " #n ::: "memory")
#else
static __device__ __forceinline__ void async_copy16(void* lds_dst, const void* gsrc) {
  *(v8h*)lds_dst = *(const v8h*)gsrc;
}
#define ASYNC_WAIT(n)
#endif

// =====================================================================
// f32 -> f16 conversion (vectorized x4)
// =====================================================================
__global__ void cvt4_kernel(const float* __restrict__ s, _Float16* __restrict__ d, long n4) {
  long t = (long)blockIdx.x * blockDim.x + threadIdx.x;
  if (t >= n4) return;
  float4 f = ((const float4*)s)[t];
  v4h h = { (_Float16)f.x, (_Float16)f.y, (_Float16)f.z, (_Float16)f.w };
  ((v4h*)d)[t] = h;
}

// =====================================================================
// Tiled WMMA GEMM:  C[M,Nn] = act(A[M,K] @ B[Nn,K]^T + bias)
// A,B are f16 in memory. Double-buffered LDS staging with
// global_load_async_to_lds_b128: tile s+1 streams in (ASYNCcnt 8..16)
// while tile s is consumed by 16 v_wmma_f32_16x16x32_f16.
// Block tile 128x128x64, 8 waves, each wave owns a 32x64 C tile.
// K%64==0, Nn%128==0. M tail handled by clamping the row index
// (C row m depends only on A row m; garbage rows are never stored).
// =====================================================================
#define BM 128
#define BN 128
#define BK 64
#define LDH 72  // padded f16 stride per LDS row (144 B, 16B-multiple)

static __device__ __forceinline__ void stage_tiles(
    _Float16* __restrict__ As, _Float16* __restrict__ Bs,
    const _Float16* __restrict__ A, const _Float16* __restrict__ B,
    int m0, int n0, int M, int K, int k0, int tid) {
#pragma unroll
  for (int p = 0; p < 4; ++p) {
    int idx = p * 256 + tid;
    int r = idx >> 3;  // 0..127 tile row
    int c = idx & 7;   // 16B chunk within 64-half row
    int gm = m0 + r; if (gm >= M) gm = M - 1;  // clamp tail
    async_copy16(&As[r * LDH + c * 8], A + (size_t)gm * K + k0 + c * 8);
    int gn = n0 + r;                           // Nn%128==0 -> in range
    async_copy16(&Bs[r * LDH + c * 8], B + (size_t)gn * K + k0 + c * 8);
  }
}

__global__ __launch_bounds__(256)
void gemm_h16_wmma(const _Float16* __restrict__ A, const _Float16* __restrict__ B,
                   const float* __restrict__ bias, float* __restrict__ Cf,
                   _Float16* __restrict__ Ch, int M, int K, int Nn, int relu) {
  __shared__ _Float16 As[2][BM * LDH];
  __shared__ _Float16 Bs[2][BN * LDH];

  const int tid  = threadIdx.x;
  const int lane = tid & 31;
  const int wid  = tid >> 5;
  const int wm   = wid & 3;   // wave M offset = wm*32
  const int wn   = wid >> 2;  // wave N offset = wn*64
  const int m0   = blockIdx.y * BM;
  const int n0   = blockIdx.x * BN;

  v8f acc[2][4];
#pragma unroll
  for (int i = 0; i < 2; ++i)
#pragma unroll
    for (int j = 0; j < 4; ++j) acc[i][j] = (v8f){0,0,0,0,0,0,0,0};

  const int nsteps = K / BK;
  // prologue: start streaming tile 0
  stage_tiles(As[0], Bs[0], A, B, m0, n0, M, K, 0, tid);

  for (int s = 0; s < nsteps; ++s) {
    const int cur = s & 1;
    if (s + 1 < nsteps) {
      // kick off tile s+1 into the other buffer, then wait only for tile s
      stage_tiles(As[cur ^ 1], Bs[cur ^ 1], A, B, m0, n0, M, K, (s + 1) * BK, tid);
      ASYNC_WAIT(8);
    } else {
      ASYNC_WAIT(0);
    }
    __syncthreads();

    const _Float16* Asb = As[cur];
    const _Float16* Bsb = Bs[cur];
    // ---- two K=32 substeps; fragments per ISA 16-bit layouts ----
#pragma unroll
    for (int ks = 0; ks < 2; ++ks) {
      v16h afrag[2];
#pragma unroll
      for (int tm = 0; tm < 2; ++tm) {
        int mrow = wm * 32 + tm * 16 + (lane & 15);
        int kb = (lane >> 4) * 8;
        v8h lo = *(const v8h*)(&Asb[mrow * LDH + ks * 32 + kb]);
        v8h hi = *(const v8h*)(&Asb[mrow * LDH + ks * 32 + kb + 16]);
        afrag[tm] = cat8(lo, hi);
      }
      v16h bfrag[4];
#pragma unroll
      for (int tn = 0; tn < 4; ++tn) {
        int nrow = wn * 64 + tn * 16 + (lane & 15);
        int kb = (lane >> 4) * 8;
        v8h lo = *(const v8h*)(&Bsb[nrow * LDH + ks * 32 + kb]);
        v8h hi = *(const v8h*)(&Bsb[nrow * LDH + ks * 32 + kb + 16]);
        bfrag[tn] = cat8(lo, hi);
      }
#pragma unroll
      for (int tm = 0; tm < 2; ++tm)
#pragma unroll
        for (int tn = 0; tn < 4; ++tn)
          acc[tm][tn] = __builtin_amdgcn_wmma_f32_16x16x32_f16(
              false, afrag[tm], false, bfrag[tn], (short)0, acc[tm][tn], false, false);
    }
    __syncthreads();  // all reads of this buffer retired before it is restaged
  }

  // ---- epilogue: bias + optional ReLU; fp32 and/or f16 outputs ----
#pragma unroll
  for (int tm = 0; tm < 2; ++tm) {
    int mbase = m0 + wm * 32 + tm * 16 + (lane >> 4) * 8;
#pragma unroll
    for (int tn = 0; tn < 4; ++tn) {
      int gn = n0 + wn * 64 + tn * 16 + (lane & 15);
      float bb = bias ? bias[gn] : 0.0f;
#pragma unroll
      for (int r = 0; r < 8; ++r) {
        int gm = mbase + r;
        if (gm < M) {
          float v = acc[tm][tn][r] + bb;
          if (relu) v = v > 0.0f ? v : 0.0f;
          if (Cf) Cf[(size_t)gm * Nn + gn] = v;
          if (Ch) Ch[(size_t)gm * Nn + gn] = (_Float16)v;
        }
      }
    }
  }
}

// =====================================================================
// GAT helper kernels
// =====================================================================
__global__ void score_kernel(const float* __restrict__ h, const float* __restrict__ a1,
                             const float* __restrict__ a2, float* __restrict__ s1o,
                             float* __restrict__ s2o, int n) {
  int w = (int)((blockIdx.x * blockDim.x + threadIdx.x) >> 5);
  int lane = threadIdx.x & 31;
  if (w >= n) return;
  const float* row = h + (size_t)w * DMODEL + lane * 8;
  const float* u1 = a1 + lane * 8;
  const float* u2 = a2 + lane * 8;
  float p1 = 0.f, p2 = 0.f;
#pragma unroll
  for (int i = 0; i < 8; ++i) { float v = row[i]; p1 += v * u1[i]; p2 += v * u2[i]; }
  p1 = wave_sum(p1); p2 = wave_sum(p2);
  if (lane == 0) { s1o[w] = p1; s2o[w] = p2; }
}

__global__ void gat_init_kernel(float* __restrict__ agg, float* __restrict__ m,
                                float* __restrict__ den, int n) {
  int t = blockIdx.x * blockDim.x + threadIdx.x;
  if (t < n * DMODEL) agg[t] = 0.0f;
  if (t < n) { m[t] = -INFINITY; den[t] = 0.0f; }
}

static __device__ __forceinline__ void edge_endpoints(const int* ei, int E, int e,
                                                      int& s, int& d) {
  if (e < E) { s = ei[e]; d = ei[E + e]; }
  else       { s = d = e - E; }  // self loops
}

__global__ void edge_max_kernel(const int* __restrict__ ei, int E, int n,
                                const float* __restrict__ ssrc, const float* __restrict__ sdst,
                                float* __restrict__ ebuf, float* __restrict__ m) {
  int e = blockIdx.x * blockDim.x + threadIdx.x;
  if (e >= E + n) return;
  int s, d; edge_endpoints(ei, E, e, s, d);
  float sc = ssrc[s] + sdst[d];
  sc = sc > 0.0f ? sc : 0.2f * sc;  // LeakyReLU(0.2)
  ebuf[e] = sc;
  atomicMaxF(&m[d], sc);
}

__global__ void edge_exp_kernel(const int* __restrict__ ei, int E, int n,
                                float* __restrict__ ebuf, const float* __restrict__ m,
                                float* __restrict__ den) {
  int e = blockIdx.x * blockDim.x + threadIdx.x;
  if (e >= E + n) return;
  int s, d; edge_endpoints(ei, E, e, s, d);
  float ex = __expf(ebuf[e] - m[d]);
  ebuf[e] = ex;
  atomicAdd(&den[d], ex);
}

__global__ void edge_agg_kernel(const int* __restrict__ ei, int E, int n,
                                const float* __restrict__ ebuf, const float* __restrict__ h,
                                float* __restrict__ agg) {
  long t = (long)blockIdx.x * blockDim.x + threadIdx.x;
  int e = (int)(t >> 5);
  int lane = (int)(t & 31);
  if (e >= E + n) return;
  int s, d; edge_endpoints(ei, E, e, s, d);
  float ex = ebuf[e];
  const float4* hp = (const float4*)(h + (size_t)s * DMODEL + lane * 8);
  float4 a = hp[0], b = hp[1];
  float* op = agg + (size_t)d * DMODEL + lane * 8;
  atomicAdd(op + 0, ex * a.x); atomicAdd(op + 1, ex * a.y);
  atomicAdd(op + 2, ex * a.z); atomicAdd(op + 3, ex * a.w);
  atomicAdd(op + 4, ex * b.x); atomicAdd(op + 5, ex * b.y);
  atomicAdd(op + 6, ex * b.z); atomicAdd(op + 7, ex * b.w);
}

__global__ void gat_final_kernel(const float* __restrict__ agg, const float* __restrict__ den,
                                 const float* __restrict__ bias, float* __restrict__ xo,
                                 _Float16* __restrict__ xo16, int n) {
  int t = blockIdx.x * blockDim.x + threadIdx.x;
  if (t >= n * DMODEL) return;
  int node = t >> 8, c = t & 255;
  float v = agg[t] / den[node] + bias[c];
  v = v > 0.0f ? v : 0.0f;  // ReLU
  xo[t] = v;
  if (xo16) xo16[t] = (_Float16)v;
}

// =====================================================================
// sequence assembly, attention, layernorm
// =====================================================================
__global__ void seq_kernel(const float* __restrict__ x1, const float* __restrict__ x2,
                           const float* __restrict__ cls, const float* __restrict__ pos,
                           float* __restrict__ seq, _Float16* __restrict__ seqh, int n) {
  int t = blockIdx.x * blockDim.x + threadIdx.x;
  if (t >= n * DMODEL) return;
  int node = t >> 8, c = t & 255;
  size_t b = (size_t)node * 3 * DMODEL;
  float s0 = cls[c] + pos[c];
  float s1 = x1[t]  + pos[DMODEL + c];
  float s2 = x2[t]  + pos[2 * DMODEL + c];
  seq[b + c] = s0;            seqh[b + c] = (_Float16)s0;
  seq[b + DMODEL + c] = s1;   seqh[b + DMODEL + c] = (_Float16)s1;
  seq[b + 2*DMODEL + c] = s2; seqh[b + 2*DMODEL + c] = (_Float16)s2;
}

// one wave per (node, head): S=3, dh=64 -> 2 channels per lane; f16 output
__global__ void attn_kernel(const float* __restrict__ qkv, _Float16* __restrict__ out, int n) {
  int gw = (int)((blockIdx.x * blockDim.x + threadIdx.x) >> 5);
  int lane = threadIdx.x & 31;
  if (gw >= n * HEADS) return;
  int node = gw >> 2, head = gw & 3;
  const float* base = qkv + (size_t)node * 3 * 768 + head * 64 + lane * 2;
  float2 q[3], k[3], v[3];
#pragma unroll
  for (int s = 0; s < 3; ++s) {
    q[s] = *(const float2*)(base + s * 768);
    k[s] = *(const float2*)(base + s * 768 + 256);
    v[s] = *(const float2*)(base + s * 768 + 512);
  }
  float lg[3][3];
#pragma unroll
  for (int i = 0; i < 3; ++i)
#pragma unroll
    for (int j = 0; j < 3; ++j)
      lg[i][j] = wave_sum(q[i].x * k[j].x + q[i].y * k[j].y) * 0.125f;  // 1/sqrt(64)

  _Float16* obase = out + (size_t)node * 3 * DMODEL + head * 64 + lane * 2;
#pragma unroll
  for (int i = 0; i < 3; ++i) {
    float mx = fmaxf(fmaxf(lg[i][0], lg[i][1]), lg[i][2]);
    float e0 = __expf(lg[i][0] - mx), e1 = __expf(lg[i][1] - mx), e2 = __expf(lg[i][2] - mx);
    float inv = 1.0f / (e0 + e1 + e2);
    float ox = (e0 * v[0].x + e1 * v[1].x + e2 * v[2].x) * inv;
    float oy = (e0 * v[0].y + e1 * v[1].y + e2 * v[2].y) * inv;
    v2h o2 = { (_Float16)ox, (_Float16)oy };
    *(v2h*)(obase + (size_t)i * DMODEL) = o2;
  }
}

// wave per row: seq[row] = LN(seq[row] + res[row]) * g + b, in place, + f16 copy
__global__ void ln_res_kernel(float* __restrict__ seq, const float* __restrict__ res,
                              const float* __restrict__ g, const float* __restrict__ bta,
                              _Float16* __restrict__ seqh, int rows) {
  int w = (int)((blockIdx.x * blockDim.x + threadIdx.x) >> 5);
  int lane = threadIdx.x & 31;
  if (w >= rows) return;
  float* row = seq + (size_t)w * DMODEL + lane * 8;
  const float* rr = res + (size_t)w * DMODEL + lane * 8;
  float4 a0 = ((const float4*)row)[0], a1 = ((const float4*)row)[1];
  float4 r0 = ((const float4*)rr)[0], r1 = ((const float4*)rr)[1];
  float x[8] = { a0.x + r0.x, a0.y + r0.y, a0.z + r0.z, a0.w + r0.w,
                 a1.x + r1.x, a1.y + r1.y, a1.z + r1.z, a1.w + r1.w };
  float s = 0.f, s2 = 0.f;
#pragma unroll
  for (int i = 0; i < 8; ++i) { s += x[i]; s2 += x[i] * x[i]; }
  s = wave_sum(s); s2 = wave_sum(s2);
  float mean = s * (1.0f / DMODEL);
  float var  = s2 * (1.0f / DMODEL) - mean * mean;
  float inv  = rsqrtf(var + 1e-5f);
  int c0 = lane * 8;
  float y[8];
#pragma unroll
  for (int i = 0; i < 8; ++i) y[i] = (x[i] - mean) * inv * g[c0 + i] + bta[c0 + i];
  ((float4*)row)[0] = make_float4(y[0], y[1], y[2], y[3]);
  ((float4*)row)[1] = make_float4(y[4], y[5], y[6], y[7]);
  _Float16* hrow = seqh + (size_t)w * DMODEL + c0;
  v4h h0 = { (_Float16)y[0], (_Float16)y[1], (_Float16)y[2], (_Float16)y[3] };
  v4h h1 = { (_Float16)y[4], (_Float16)y[5], (_Float16)y[6], (_Float16)y[7] };
  ((v4h*)hrow)[0] = h0;
  ((v4h*)hrow)[1] = h1;
}

// final LN on token 0 only -> d_out[n, 256]
__global__ void final_ln_kernel(const float* __restrict__ seq, const float* __restrict__ g,
                                const float* __restrict__ bta, float* __restrict__ out, int n) {
  int w = (int)((blockIdx.x * blockDim.x + threadIdx.x) >> 5);
  int lane = threadIdx.x & 31;
  if (w >= n) return;
  const float* row = seq + (size_t)w * 3 * DMODEL + lane * 8;
  float4 a0 = ((const float4*)row)[0], a1 = ((const float4*)row)[1];
  float x[8] = { a0.x, a0.y, a0.z, a0.w, a1.x, a1.y, a1.z, a1.w };
  float s = 0.f, s2 = 0.f;
#pragma unroll
  for (int i = 0; i < 8; ++i) { s += x[i]; s2 += x[i] * x[i]; }
  s = wave_sum(s); s2 = wave_sum(s2);
  float mean = s * (1.0f / DMODEL);
  float var  = s2 * (1.0f / DMODEL) - mean * mean;
  float inv  = rsqrtf(var + 1e-5f);
  int c0 = lane * 8;
  float* op = out + (size_t)w * DMODEL + c0;
  float y[8];
#pragma unroll
  for (int i = 0; i < 8; ++i) y[i] = (x[i] - mean) * inv * g[c0 + i] + bta[c0 + i];
  ((float4*)op)[0] = make_float4(y[0], y[1], y[2], y[3]);
  ((float4*)op)[1] = make_float4(y[4], y[5], y[6], y[7]);
}

// =====================================================================
// host launch
// =====================================================================
extern "C" void kernel_launch(void* const* d_in, const int* in_sizes, int n_in,
                              void* d_out, int out_size, void* d_ws, size_t ws_size,
                              hipStream_t stream) {
  (void)in_sizes; (void)n_in; (void)out_size; (void)ws_size;

  const float* x    = (const float*)d_in[0];
  const int*   ei   = (const int*)d_in[1];
  const float* g1W  = (const float*)d_in[2];
  const float* g1b  = (const float*)d_in[3];
  const float* g1as = (const float*)d_in[4];
  const float* g1ad = (const float*)d_in[5];
  const float* g2W  = (const float*)d_in[6];
  const float* g2b  = (const float*)d_in[7];
  const float* g2as = (const float*)d_in[8];
  const float* g2ad = (const float*)d_in[9];
  const float* cls  = (const float*)d_in[10];
  const float* pos  = (const float*)d_in[11];
  const float* Wqkv = (const float*)d_in[12];
  const float* bqkv = (const float*)d_in[13];
  const float* Wo   = (const float*)d_in[14];
  const float* bo   = (const float*)d_in[15];
  const float* ln1g = (const float*)d_in[16];
  const float* ln1b = (const float*)d_in[17];
  const float* ln2g = (const float*)d_in[18];
  const float* ln2b = (const float*)d_in[19];
  const float* Wff1 = (const float*)d_in[20];
  const float* bff1 = (const float*)d_in[21];
  const float* Wff2 = (const float*)d_in[22];
  const float* bff2 = (const float*)d_in[23];
  const float* ng   = (const float*)d_in[24];
  const float* nb   = (const float*)d_in[25];
  float* out = (float*)d_out;

  const int NN = N_NODES, EE = E_EDGES, Etot = EE + NN;

  // ---- fp32 workspace region ----
  float* ws = (float*)d_ws;
  size_t o = 0;
  auto alloc_f = [&](size_t n) { float* p = ws + o; o += (n + 63) & ~(size_t)63; return p; };
  float* h    = alloc_f((size_t)NN * DMODEL);
  float* agg  = alloc_f((size_t)NN * DMODEL);
  float* x1   = alloc_f((size_t)NN * DMODEL);
  float* x2   = alloc_f((size_t)NN * DMODEL);
  float* seq  = alloc_f((size_t)NN * 3 * DMODEL);
  float* tmp  = alloc_f((size_t)NN * 3 * DMODEL);
  float* qkvf = alloc_f((size_t)NN * 3 * 768);
  float* ssrc = alloc_f(NN);
  float* sdst = alloc_f(NN);
  float* mbuf = alloc_f(NN);
  float* den  = alloc_f(NN);
  float* ebuf = alloc_f(Etot);

  // ---- f16 workspace region ----
  _Float16* wh = (_Float16*)(ws + o);
  size_t oh = 0;
  auto alloc_h = [&](size_t n) { _Float16* p = wh + oh; oh += (n + 63) & ~(size_t)63; return p; };
  _Float16* xh    = alloc_h((size_t)NN * DIN);
  _Float16* x1h   = alloc_h((size_t)NN * DMODEL);
  _Float16* seqh  = alloc_h((size_t)NN * 3 * DMODEL);
  _Float16* attnh = alloc_h((size_t)NN * 3 * DMODEL);
  _Float16* Uh    = alloc_h((size_t)NN * 3 * 1024);
  _Float16* g1Wh  = alloc_h((size_t)DMODEL * DIN);
  _Float16* g2Wh  = alloc_h((size_t)DMODEL * DMODEL);
  _Float16* Wqkvh = alloc_h((size_t)NLAYERS * 768 * 256);
  _Float16* Woh   = alloc_h((size_t)NLAYERS * 256 * 256);
  _Float16* Wff1h = alloc_h((size_t)NLAYERS * 1024 * 256);
  _Float16* Wff2h = alloc_h((size_t)NLAYERS * 256 * 1024);

  const int TB = 256;
  auto cvt = [&](const float* s, _Float16* d, long n) {
    long n4 = n / 4;
    cvt4_kernel<<<(int)((n4 + TB - 1) / TB), TB, 0, stream>>>(s, d, n4);
  };
  auto gemm = [&](const _Float16* A, const _Float16* B, const float* bias,
                  float* Cf, _Float16* Ch, int M, int K, int Nn, int relu) {
    dim3 grid(Nn / 128, (M + 127) / 128);
    gemm_h16_wmma<<<grid, 256, 0, stream>>>(A, B, bias, Cf, Ch, M, K, Nn, relu);
  };

  const int blocksNC    = (NN * DMODEL + TB - 1) / TB;
  const int blocksE     = (Etot + TB - 1) / TB;
  const int blocksAgg   = (int)(((size_t)Etot * 32 + TB - 1) / TB);
  const int blocksWaveN = (NN * 32 + TB - 1) / TB;
  const int blocksRows3 = (NN * 3 * 32 + TB - 1) / TB;
  const int blocksAttn  = (NN * HEADS * 32 + TB - 1) / TB;

  // ---------------- one-time f16 conversions ----------------
  cvt(x, xh, (long)NN * DIN);
  cvt(g1W, g1Wh, (long)DMODEL * DIN);
  cvt(g2W, g2Wh, (long)DMODEL * DMODEL);
  cvt(Wqkv, Wqkvh, (long)NLAYERS * 768 * 256);
  cvt(Wo, Woh, (long)NLAYERS * 256 * 256);
  cvt(Wff1, Wff1h, (long)NLAYERS * 1024 * 256);
  cvt(Wff2, Wff2h, (long)NLAYERS * 256 * 1024);

  // ---------------- GAT layer 1 ----------------
  gemm(xh, g1Wh, nullptr, h, nullptr, NN, DIN, DMODEL, 0);
  score_kernel<<<blocksWaveN, TB, 0, stream>>>(h, g1as, g1ad, ssrc, sdst, NN);
  gat_init_kernel<<<blocksNC, TB, 0, stream>>>(agg, mbuf, den, NN);
  edge_max_kernel<<<blocksE, TB, 0, stream>>>(ei, EE, NN, ssrc, sdst, ebuf, mbuf);
  edge_exp_kernel<<<blocksE, TB, 0, stream>>>(ei, EE, NN, ebuf, mbuf, den);
  edge_agg_kernel<<<blocksAgg, TB, 0, stream>>>(ei, EE, NN, ebuf, h, agg);
  gat_final_kernel<<<blocksNC, TB, 0, stream>>>(agg, den, g1b, x1, x1h, NN);

  // ---------------- GAT layer 2 ----------------
  gemm(x1h, g2Wh, nullptr, h, nullptr, NN, DMODEL, DMODEL, 0);
  score_kernel<<<blocksWaveN, TB, 0, stream>>>(h, g2as, g2ad, ssrc, sdst, NN);
  gat_init_kernel<<<blocksNC, TB, 0, stream>>>(agg, mbuf, den, NN);
  edge_max_kernel<<<blocksE, TB, 0, stream>>>(ei, EE, NN, ssrc, sdst, ebuf, mbuf);
  edge_exp_kernel<<<blocksE, TB, 0, stream>>>(ei, EE, NN, ebuf, mbuf, den);
  edge_agg_kernel<<<blocksAgg, TB, 0, stream>>>(ei, EE, NN, ebuf, h, agg);
  gat_final_kernel<<<blocksNC, TB, 0, stream>>>(agg, den, g2b, x2, nullptr, NN);

  // ---------------- sequence assembly ----------------
  seq_kernel<<<blocksNC, TB, 0, stream>>>(x1, x2, cls, pos, seq, seqh, NN);

  // ---------------- transformer layers ----------------
  for (int l = 0; l < NLAYERS; ++l) {
    gemm(seqh, Wqkvh + (size_t)l * 768 * 256, bqkv + l * 768, qkvf, nullptr,
         NN * 3, 256, 768, 0);
    attn_kernel<<<blocksAttn, TB, 0, stream>>>(qkvf, attnh, NN);
    gemm(attnh, Woh + (size_t)l * 256 * 256, bo + l * 256, tmp, nullptr,
         NN * 3, 256, 256, 0);
    ln_res_kernel<<<blocksRows3, TB, 0, stream>>>(seq, tmp, ln1g + l * 256, ln1b + l * 256,
                                                  seqh, NN * 3);
    gemm(seqh, Wff1h + (size_t)l * 1024 * 256, bff1 + l * 1024, nullptr, Uh,
         NN * 3, 256, 1024, 1);
    gemm(Uh, Wff2h + (size_t)l * 256 * 1024, bff2 + l * 256, tmp, nullptr,
         NN * 3, 1024, 256, 0);
    ln_res_kernel<<<blocksRows3, TB, 0, stream>>>(seq, tmp, ln2g + l * 256, ln2b + l * 256,
                                                  seqh, NN * 3);
  }

  // ---------------- final LN on token 0 ----------------
  final_ln_kernel<<<blocksWaveN, TB, 0, stream>>>(seq, ng, nb, out, NN);
}
